// CAE2Layer_23785528885730
// MI455X (gfx1250) — compile-verified, matrix operand order checked by reading, not run on hardware
//
#include <hip/hip_runtime.h>

// ---------------------------------------------------------------------------
// CDNA5 (gfx1250) contractive-autoencoder forward + Jacobian.
// GEMMs on v_wmma_f32_16x16x32_bf16 (fp32 accumulate).
// Jacobian kernel stages its B tiles through LDS via the Tensor Data Mover
// (tensor_load_to_lds, TENSORcnt, double-buffered) per CDNA5 ISA.
// ---------------------------------------------------------------------------

typedef __attribute__((ext_vector_type(16))) __bf16 v16bf;
typedef __attribute__((ext_vector_type(8)))  __bf16 v8bf;
typedef __attribute__((ext_vector_type(8)))  float  v8f;
typedef __attribute__((ext_vector_type(4)))  unsigned int u32x4;
typedef __attribute__((ext_vector_type(8)))  int i32x8;
typedef __attribute__((ext_vector_type(4)))  int i32x4;

constexpr int Bn  = 512;   // batch
constexpr int Dn  = 1024;  // input dim
constexpr int H1n = 512;   // hidden 1
constexpr int H2n = 128;   // hidden 2 (code)

__device__ __forceinline__ float sigmoidf(float v) {
  return 1.0f / (1.0f + __expf(-v));
}

// Load one 16x32 bf16 fragment (A-layout, ISA 7.12.2) from a row-major
// [rows, K] matrix: lane m=lane&15 -> row, half=lane>>4 selects K sub-blocks.
__device__ __forceinline__ v16bf load_frag(const __bf16* __restrict__ base,
                                           int ld, int row, int k, int lane) {
  const int m = lane & 15;
  const int half = lane >> 4;
  const __bf16* p = base + (size_t)(row + m) * ld + k + half * 8;
  v8bf lo = *(const v8bf*)p;          // K = k + half*8 + [0..7]
  v8bf hi = *(const v8bf*)(p + 16);   // K = k + 16 + half*8 + [0..7]
  return __builtin_shufflevector(lo, hi, 0, 1, 2, 3, 4, 5, 6, 7,
                                         8, 9, 10, 11, 12, 13, 14, 15);
}

// Same fragment gather from an LDS slab row: p0 points at (row, half*8).
__device__ __forceinline__ v16bf load_frag_pair(const __bf16* p0) {
  v8bf lo = *(const v8bf*)p0;
  v8bf hi = *(const v8bf*)(p0 + 16);
  return __builtin_shufflevector(lo, hi, 0, 1, 2, 3, 4, 5, 6, 7,
                                         8, 9, 10, 11, 12, 13, 14, 15);
}

__device__ __forceinline__ v8f wmma_bf16(v16bf a, v16bf b, v8f c) {
  return __builtin_amdgcn_wmma_f32_16x16x32_bf16(false, a, false, b,
                                                 (short)0, c, false, false);
}

// 32x64 wave tile (2 M-frags x 4 N-frags), K-loop over bf16 WMMA (K=32 step).
// A: row-major [M,K]; Bt: row-major [N,K].
__device__ __forceinline__ void gemm_core(const __bf16* __restrict__ A, int lda,
                                          const __bf16* __restrict__ Bt, int ldb,
                                          int m0, int n0, int K, v8f acc[2][4]) {
  const int lane = threadIdx.x & 31;
  for (int k = 0; k < K; k += 32) {
    v16bf a0 = load_frag(A, lda, m0, k, lane);
    v16bf a1 = load_frag(A, lda, m0 + 16, k, lane);
    v16bf b0 = load_frag(Bt, ldb, n0, k, lane);
    v16bf b1 = load_frag(Bt, ldb, n0 + 16, k, lane);
    v16bf b2 = load_frag(Bt, ldb, n0 + 32, k, lane);
    v16bf b3 = load_frag(Bt, ldb, n0 + 48, k, lane);
    acc[0][0] = wmma_bf16(a0, b0, acc[0][0]);
    acc[0][1] = wmma_bf16(a0, b1, acc[0][1]);
    acc[0][2] = wmma_bf16(a0, b2, acc[0][2]);
    acc[0][3] = wmma_bf16(a0, b3, acc[0][3]);
    acc[1][0] = wmma_bf16(a1, b0, acc[1][0]);
    acc[1][1] = wmma_bf16(a1, b1, acc[1][1]);
    acc[1][2] = wmma_bf16(a1, b2, acc[1][2]);
    acc[1][3] = wmma_bf16(a1, b3, acc[1][3]);
  }
}

// Iterate the 64 fp32 accumulator elements this lane owns (C layout 7.12.2).
template <typename F>
__device__ __forceinline__ void for_each_elem(const v8f acc[2][4], int m0, int n0,
                                              F&& f) {
  const int lane = threadIdx.x & 31;
  const int nl = lane & 15;
  const int mh = (lane >> 4) * 8;
#pragma unroll
  for (int i = 0; i < 2; ++i)
#pragma unroll
    for (int j = 0; j < 4; ++j)
#pragma unroll
      for (int r = 0; r < 8; ++r)
        f(m0 + 16 * i + mh + r, n0 + 16 * j + nl, acc[i][j][r]);
}

__device__ __forceinline__ void tile_origin(int& m0, int& n0) {
  const int wave = threadIdx.x >> 5;  // 8 waves: 4 (M) x 2 (N)
  m0 = blockIdx.x * 128 + (wave & 3) * 32;
  n0 = blockIdx.y * 128 + (wave >> 2) * 64;
}

#define ZERO_ACC(acc)                                   \
  v8f zz_ = {0.f, 0.f, 0.f, 0.f, 0.f, 0.f, 0.f, 0.f};  \
  v8f acc[2][4] = {{zz_, zz_, zz_, zz_}, {zz_, zz_, zz_, zz_}}

// ---------------------------------------------------------------------------
// Tensor Data Mover: 2D tile load Global -> LDS (ISA 08, D# groups 0/1).
// data_size = 2 bytes; LDS row padding: 16 DWORDs data + 4 DWORDs pad
// (pad_interval code 3 = 64B, pad_amount code 3 = 16B) -> 80B LDS row pitch.
// This toolchain exposes the 6-arg builtin:
//   (u32x4 g0, i32x8 g1, i32x4 g2, i32x4 g3, i32x8 extra, i32 cpol)
// ---------------------------------------------------------------------------
__device__ __forceinline__ void tdm_load_2d_bf16(
    unsigned int lds_byte_off, const void* gaddr, unsigned int tile_d0,
    unsigned int tile_d1, unsigned int tensor_d0, unsigned int tensor_d1,
    unsigned int stride0) {
  unsigned long long ga = (unsigned long long)(uintptr_t)gaddr;
  u32x4 g0;
  g0[0] = 1u;                                   // count=1, no gather
  g0[1] = lds_byte_off;                         // lds_addr
  g0[2] = (unsigned int)ga;                     // global_addr[31:0]
  g0[3] = (unsigned int)(ga >> 32) | 0x80000000u;  // addr[56:32] | type=2
  i32x8 g1;
  g1[0] = (int)((1u << 16)       // data_size = 2 bytes
                | (1u << 20)     // pad_enable
                | (3u << 22)     // pad_interval: 64B of data
                | (3u << 25));   // pad_amount: 16B of pad
  g1[1] = (int)(tensor_d0 << 16);                       // tensor_dim0[15:0]
  g1[2] = (int)((tensor_d0 >> 16) | (tensor_d1 << 16)); // dim0 hi | dim1 lo
  g1[3] = (int)((tensor_d1 >> 16) | (tile_d0 << 16));   // dim1 hi | tile_dim0
  g1[4] = (int)(tile_d1 & 0xFFFFu);                     // tile_dim1, tile_dim2=0
  g1[5] = (int)stride0;                                 // tensor_dim0_stride lo
  g1[6] = 0;
  g1[7] = 0;
  i32x4 z4 = {0, 0, 0, 0};
  i32x8 z8 = {0, 0, 0, 0, 0, 0, 0, 0};
  __builtin_amdgcn_tensor_load_to_lds(g0, g1, z4, z4, z8, 0);
}

// ------------------------- prep: fp32 -> bf16 ------------------------------
__global__ void k_cvt(const float* __restrict__ src, __bf16* __restrict__ dst,
                      int n) {
  int i = blockIdx.x * blockDim.x + threadIdx.x;
  if (i < n) dst[i] = (__bf16)src[i];
}

__global__ void k_transpose_cvt(const float* __restrict__ src,
                                __bf16* __restrict__ dst, int rows, int cols) {
  int c = blockIdx.x * 16 + threadIdx.x;
  int r = blockIdx.y * 16 + threadIdx.y;
  if (r < rows && c < cols)
    dst[(size_t)c * rows + r] = (__bf16)src[(size_t)r * cols + c];
}

// ------------------------- encoder layer 1 ---------------------------------
__global__ void k_enc1(const __bf16* __restrict__ x_bf,
                       const __bf16* __restrict__ W1_bf,
                       const float* __restrict__ b1,
                       __bf16* __restrict__ c1_bf,
                       __bf16* __restrict__ s1p_bf) {
  int m0, n0;
  tile_origin(m0, n0);
  ZERO_ACC(acc);
  gemm_core(x_bf, Dn, W1_bf, Dn, m0, n0, Dn, acc);
  for_each_elem(acc, m0, n0, [&](int m, int n, float v) {
    float s = sigmoidf(v + b1[n]);
    c1_bf[(size_t)m * H1n + n] = (__bf16)s;
    s1p_bf[(size_t)m * H1n + n] = (__bf16)(s * (1.0f - s));
  });
}

// ------------------------- encoder layer 2 ---------------------------------
__global__ void k_enc2(const __bf16* __restrict__ c1_bf,
                       const __bf16* __restrict__ W2_bf,
                       const float* __restrict__ b2,
                       float* __restrict__ out_c2,
                       __bf16* __restrict__ c2_bf,
                       float* __restrict__ s2p) {
  int m0, n0;
  tile_origin(m0, n0);
  ZERO_ACC(acc);
  gemm_core(c1_bf, H1n, W2_bf, H1n, m0, n0, H1n, acc);
  for_each_elem(acc, m0, n0, [&](int m, int n, float v) {
    float s = sigmoidf(v + b2[n]);
    out_c2[(size_t)m * H2n + n] = s;
    c2_bf[(size_t)m * H2n + n] = (__bf16)s;
    s2p[(size_t)m * H2n + n] = s * (1.0f - s);
  });
}

// ------------------------- decoder layer 1 ---------------------------------
__global__ void k_dec1(const __bf16* __restrict__ c2_bf,
                       const __bf16* __restrict__ W2T_bf,
                       const float* __restrict__ b3,
                       __bf16* __restrict__ c3_bf) {
  int m0, n0;
  tile_origin(m0, n0);
  ZERO_ACC(acc);
  gemm_core(c2_bf, H2n, W2T_bf, H2n, m0, n0, H2n, acc);
  for_each_elem(acc, m0, n0, [&](int m, int n, float v) {
    c3_bf[(size_t)m * H1n + n] = (__bf16)sigmoidf(v + b3[n]);
  });
}

// ------------------------- decoder layer 2 ---------------------------------
__global__ void k_dec2(const __bf16* __restrict__ c3_bf,
                       const __bf16* __restrict__ W1T_bf,
                       const float* __restrict__ br,
                       float* __restrict__ out_rec) {
  int m0, n0;
  tile_origin(m0, n0);
  ZERO_ACC(acc);
  gemm_core(c3_bf, H1n, W1T_bf, H1n, m0, n0, H1n, acc);
  for_each_elem(acc, m0, n0, [&](int m, int n, float v) {
    out_rec[(size_t)m * Dn + n] = v + br[n];
  });
}

// ------------------------- Jacobian ----------------------------------------
// Jac[b,h,d] = s2p[b,h] * sum_k (W2[h,k]*s1p[b,k]) * W1[k,d]
// Per workgroup: one batch b, one 128-wide d tile. M=128 (h), K=512, N=128.
// B slabs ([32 k x 128 d] of W1^T) are DMA'd to LDS by the TDM, double
// buffered on TENSORcnt; A fragments built in registers (W2 frag * s1p).
constexpr int KSTEPS  = H1n / 32;       // 16
constexpr int LDSROW  = 40;             // 32 data + 8 pad bf16 -> 80B pitch
constexpr int SLABSZ  = 128 * LDSROW;   // elements per buffer

__global__ void k_jac(const __bf16* __restrict__ W2_bf,
                      const __bf16* __restrict__ W1T_bf,
                      const __bf16* __restrict__ s1p_bf,
                      const float* __restrict__ s2p,
                      float* __restrict__ jac) {
  __shared__ __align__(16) __bf16 ldsB[2][SLABSZ];

  const int bb = blockIdx.x;
  const int wave = threadIdx.x >> 5;
  const int lane = threadIdx.x & 31;
  const int half = lane >> 4;
  const int nl = lane & 15;
  const int m0 = (wave & 3) * 32;            // h tile within 128
  const int ncol = (wave >> 2) * 64;         // d tile within 128 (LDS row)
  const int d0 = blockIdx.y * 128;           // d tile origin (global)

  const __bf16* s1row = s1p_bf + (size_t)bb * H1n;
  const __bf16* btile = W1T_bf + (size_t)d0 * H1n;   // [128 rows x 512 k]

  // Prologue: TDM fetch of slab k=0 into buffer 0 (one op per workgroup).
  if (wave == 0) {
    tdm_load_2d_bf16((unsigned int)(uintptr_t)&ldsB[0][0], btile,
                     /*tile k*/ 32, /*tile d*/ 128,
                     /*tensor_d0*/ H1n, /*tensor_d1*/ 128, /*stride0*/ H1n);
  }

  ZERO_ACC(acc);
  for (int ks = 0; ks < KSTEPS; ++ks) {
    const int k = ks * 32;
    const int cur = ks & 1;
    if (wave == 0) {
      if (ks + 1 < KSTEPS) {
        tdm_load_2d_bf16((unsigned int)(uintptr_t)&ldsB[cur ^ 1][0],
                         btile + (k + 32), 32, 128, H1n, 128, H1n);
        __builtin_amdgcn_s_wait_tensorcnt(1);  // slab `ks` landed
      } else {
        __builtin_amdgcn_s_wait_tensorcnt(0);  // final slab landed
      }
    }
    __syncthreads();  // publish buffer `cur` to all waves

    // A fragments: W2 slab scaled elementwise by s1p[b] (bf16).
    v8bf slo = *(const v8bf*)(s1row + k + half * 8);
    v8bf shi = *(const v8bf*)(s1row + k + 16 + half * 8);
    v16bf s = __builtin_shufflevector(slo, shi, 0, 1, 2, 3, 4, 5, 6, 7,
                                                8, 9, 10, 11, 12, 13, 14, 15);
    v16bf a0 = load_frag(W2_bf, H1n, m0, k, lane);
    v16bf a1 = load_frag(W2_bf, H1n, m0 + 16, k, lane);
#pragma unroll
    for (int t = 0; t < 16; ++t) {
      a0[t] = (__bf16)((float)a0[t] * (float)s[t]);
      a1[t] = (__bf16)((float)a1[t] * (float)s[t]);
    }

    // B fragments out of the LDS slab (two ds_load_b128 each).
    const __bf16* lb = &ldsB[cur][0];
    v16bf b0 = load_frag_pair(lb + (ncol + nl) * LDSROW + half * 8);
    v16bf b1 = load_frag_pair(lb + (ncol + 16 + nl) * LDSROW + half * 8);
    v16bf b2 = load_frag_pair(lb + (ncol + 32 + nl) * LDSROW + half * 8);
    v16bf b3 = load_frag_pair(lb + (ncol + 48 + nl) * LDSROW + half * 8);

    acc[0][0] = wmma_bf16(a0, b0, acc[0][0]);
    acc[0][1] = wmma_bf16(a0, b1, acc[0][1]);
    acc[0][2] = wmma_bf16(a0, b2, acc[0][2]);
    acc[0][3] = wmma_bf16(a0, b3, acc[0][3]);
    acc[1][0] = wmma_bf16(a1, b0, acc[1][0]);
    acc[1][1] = wmma_bf16(a1, b1, acc[1][1]);
    acc[1][2] = wmma_bf16(a1, b2, acc[1][2]);
    acc[1][3] = wmma_bf16(a1, b3, acc[1][3]);

    __syncthreads();  // all reads of `cur` done before TDM may overwrite it
  }

  const int mh = half * 8;
#pragma unroll
  for (int i = 0; i < 2; ++i) {
#pragma unroll
    for (int r = 0; r < 8; ++r) {
      const int m = m0 + 16 * i + mh + r;
      const float sp = s2p[(size_t)bb * H2n + m];
      float* row = jac + ((size_t)bb * H2n + m) * Dn;
#pragma unroll
      for (int j = 0; j < 4; ++j) {
        row[d0 + ncol + 16 * j + nl] = acc[i][j][r] * sp;
      }
    }
  }
}

// ---------------------------------------------------------------------------
extern "C" void kernel_launch(void* const* d_in, const int* in_sizes, int n_in,
                              void* d_out, int out_size, void* d_ws,
                              size_t ws_size, hipStream_t stream) {
  const float* x  = (const float*)d_in[0];  // [B, D]
  const float* W1 = (const float*)d_in[1];  // [H1, D]
  const float* b1 = (const float*)d_in[2];  // [H1]
  const float* W2 = (const float*)d_in[3];  // [H2, H1]
  const float* b2 = (const float*)d_in[4];  // [H2]
  const float* b3 = (const float*)d_in[5];  // [H1]
  const float* br = (const float*)d_in[6];  // [D]

  float* out_rec = (float*)d_out;                        // [B, D]
  float* out_c2  = out_rec + (size_t)Bn * Dn;            // [B, H2]
  float* out_jac = out_c2 + (size_t)Bn * H2n;            // [B, H2, D]

  char* ws = (char*)d_ws;
  size_t off = 0;
  auto alloc = [&](size_t bytes) -> char* {
    char* p = ws + off;
    off += (bytes + 255) & ~(size_t)255;
    return p;
  };
  __bf16* x_bf   = (__bf16*)alloc((size_t)Bn * Dn * 2);
  __bf16* W1_bf  = (__bf16*)alloc((size_t)H1n * Dn * 2);
  __bf16* W1T_bf = (__bf16*)alloc((size_t)Dn * H1n * 2);
  __bf16* W2_bf  = (__bf16*)alloc((size_t)H2n * H1n * 2);
  __bf16* W2T_bf = (__bf16*)alloc((size_t)H1n * H2n * 2);
  __bf16* c1_bf  = (__bf16*)alloc((size_t)Bn * H1n * 2);
  __bf16* s1p_bf = (__bf16*)alloc((size_t)Bn * H1n * 2);
  __bf16* c2_bf  = (__bf16*)alloc((size_t)Bn * H2n * 2);
  __bf16* c3_bf  = (__bf16*)alloc((size_t)Bn * H1n * 2);
  float*  s2p    = (float*)alloc((size_t)Bn * H2n * 4);
  (void)ws_size; (void)n_in; (void)in_sizes; (void)out_size;

  // --- prep: convert / transpose weights & inputs to bf16 ---
  {
    int n = Bn * Dn;
    k_cvt<<<(n + 255) / 256, 256, 0, stream>>>(x, x_bf, n);
    n = H1n * Dn;
    k_cvt<<<(n + 255) / 256, 256, 0, stream>>>(W1, W1_bf, n);
    n = H2n * H1n;
    k_cvt<<<(n + 255) / 256, 256, 0, stream>>>(W2, W2_bf, n);
    dim3 t16(16, 16);
    k_transpose_cvt<<<dim3(Dn / 16, H1n / 16), t16, 0, stream>>>(W1, W1T_bf,
                                                                 H1n, Dn);
    k_transpose_cvt<<<dim3(H1n / 16, H2n / 16), t16, 0, stream>>>(W2, W2T_bf,
                                                                  H2n, H1n);
  }

  // --- forward pass (all WMMA) ---
  k_enc1<<<dim3(Bn / 128, H1n / 128), 256, 0, stream>>>(x_bf, W1_bf, b1, c1_bf,
                                                        s1p_bf);
  k_enc2<<<dim3(Bn / 128, H2n / 128), 256, 0, stream>>>(c1_bf, W2_bf, b2,
                                                        out_c2, c2_bf, s2p);
  k_dec1<<<dim3(Bn / 128, H1n / 128), 256, 0, stream>>>(c2_bf, W2T_bf, b3,
                                                        c3_bf);
  k_dec2<<<dim3(Bn / 128, Dn / 128), 256, 0, stream>>>(c3_bf, W1T_bf, br,
                                                       out_rec);

  // --- Jacobian: 512 batches x 8 d-tiles = 4096 workgroups, TDM-fed ---
  k_jac<<<dim3(Bn, Dn / 128), 256, 0, stream>>>(W2_bf, W1T_bf, s1p_bf, s2p,
                                                out_jac);
}
